// Decoder_39350490366589
// MI455X (gfx1250) — matrix operand order repaired and checked by reading, not compiled, not run on hardware
//
#include <hip/hip_runtime.h>
#include <stdint.h>

// ---------------------------------------------------------------------------
// Problem constants (from reference)
// ---------------------------------------------------------------------------
static constexpr int E_DIM  = 128;
static constexpr int H_DIM  = 256;
static constexpr int T_SEG  = 32;      // NUM_SEG
static constexpr int C_CH   = 8;
static constexpr int STEP   = 32;
static constexpr int SIG    = 1056;    // SIG_LEN
static constexpr int B_BATCH = 1024;

// bf16 weight layout inside d_ws (ushort elements)
static constexpr int W1_OFF   = 0;                 // 256x128 row-major (H x E)
static constexpr int W2_OFF   = 32768;             // 128x256 row-major (E x H)
static constexpr int WINV_OFF = 65536;             // 64x128  row-major (S x E)
static constexpr int WB_TOTAL = 73728;
static constexpr size_t GSIG_BYTE_OFF = (size_t)WB_TOTAL * 2;  // 147456, 16B aligned

// ---------------------------------------------------------------------------
// Types for WMMA / vector memory ops (native clang ext-vectors)
// ---------------------------------------------------------------------------
typedef __bf16 bf16_t;
typedef bf16_t v16bf __attribute__((ext_vector_type(16)));
typedef float  v8f   __attribute__((ext_vector_type(8)));
typedef float  v2f   __attribute__((ext_vector_type(2)));

union Frag {
  v16bf          v;
  unsigned short u[16];
  unsigned int   d[8];
};

__device__ __forceinline__ unsigned short f2bf(float f) {
  // round-to-nearest-even f32 -> bf16
  unsigned int x = __float_as_uint(f);
  x += 0x7FFFu + ((x >> 16) & 1u);
  return (unsigned short)(x >> 16);
}

// ---------------------------------------------------------------------------
// Kernel 0: convert weights f32 -> bf16 once into workspace
// ---------------------------------------------------------------------------
__global__ __launch_bounds__(256) void convert_weights(
    const float* __restrict__ W1, const float* __restrict__ W2,
    const float* __restrict__ Winv, unsigned short* __restrict__ wbf)
{
  int i = blockIdx.x * 256 + threadIdx.x;
  if (i < 32768) {
    wbf[W1_OFF + i] = f2bf(W1[i]);
  } else if (i < 65536) {
    wbf[i] = f2bf(W2[i - 32768]);                 // W2_OFF == 32768
  } else if (i < WB_TOTAL) {
    wbf[i] = f2bf(Winv[i - 65536]);               // WINV_OFF == 65536
  }
}

// ---------------------------------------------------------------------------
// Helper: load a 32x16 bf16 B-fragment from row-major (N x K) bf16 weights.
// Table order (matches A): elements 0..7 hold K=kh*8+(0..7); elements 8..15 hold K=16+kh*8+(0..7).
// -> lane reads bf16 [kbase + kh*8 + (0..7)] and [kbase + kh*8 + 16..23] of row n.
// Weights are hot in L2 (hit by all 8192 blocks) -> default temporal loads.
// ---------------------------------------------------------------------------
__device__ __forceinline__ Frag load_bfrag_global(
    const unsigned short* __restrict__ w, int row, int kstride, int kbase, int kh)
{
  Frag f;
  const unsigned short* q = w + (size_t)row * kstride + kbase + kh * 8;
  uint4 lo = *(const uint4*)q;
  uint4 hi = *(const uint4*)(q + 16);
  f.d[0] = lo.x; f.d[1] = lo.y; f.d[2] = lo.z; f.d[3] = lo.w;
  f.d[4] = hi.x; f.d[5] = hi.y; f.d[6] = hi.z; f.d[7] = hi.w;
  return f;
}

// Helper: load a 16x32 bf16 A-fragment from a row-major bf16 LDS tile.
// A layout: VGPR j (0..3): K=(j*2, j*2+1)+kh*8 ; VGPR j (4..7): +16.
// -> lane m reads ushorts [base + (0..7)] and [base + 16..23].
__device__ __forceinline__ Frag load_afrag_lds(const unsigned short* q)
{
  Frag f;
  uint4 lo = *(const uint4*)q;
  uint4 hi = *(const uint4*)(q + 16);
  f.d[0] = lo.x; f.d[1] = lo.y; f.d[2] = lo.z; f.d[3] = lo.w;
  f.d[4] = hi.x; f.d[5] = hi.y; f.d[6] = hi.z; f.d[7] = hi.w;
  return f;
}

// ---------------------------------------------------------------------------
// Kernel 1: fused MLP -> inverse embedding -> overlap-add -> normalize.
// One block per (b, c): 32 rows. 2 waves of 32 lanes; each wave owns 16 rows.
// ---------------------------------------------------------------------------
__global__ __launch_bounds__(64) void decoder_gemm_ola(
    const float* __restrict__ enc,  const float* __restrict__ b1,
    const float* __restrict__ b2,   const float* __restrict__ binv,
    const unsigned short* __restrict__ wbf, float* __restrict__ gsig)
{
  __shared__ unsigned short Hlds [2 * 16 * H_DIM];   // 16 KB, per-wave slices
  __shared__ unsigned short X2lds[2 * 16 * E_DIM];   //  8 KB, per-wave slices
  __shared__ float          sig  [SIG];              //  4.2 KB, shared

  const int bc   = blockIdx.x;          // b*8 + c
  const int lane = threadIdx.x & 31;
  const int w    = threadIdx.x >> 5;    // wave 0/1
  const int m    = lane & 15;           // row within 16-row tile / N column
  const int kh   = lane >> 4;           // K half select

  for (int p = threadIdx.x; p < SIG; p += 64) sig[p] = 0.0f;

  __builtin_prefetch(wbf, 0, 3);                      // global_prefetch_b8
  __builtin_prefetch(wbf + W2_OFF, 0, 3);

  // ---- A1 fragments: wave's 16 rows of encoder_output, f32 -> bf16 --------
  // encoder_output is 134 MB streamed exactly once: non-temporal loads keep
  // the 192 MB L2 free for the reused weights + intermediate signal.
  const float* arow = enc + ((size_t)bc * T_SEG + (w * 16 + m)) * E_DIM;
  Frag A1[4];
  #pragma unroll
  for (int kb = 0; kb < 4; ++kb) {
    #pragma unroll
    for (int j = 0; j < 8; ++j) {
      int e = kb * 32 + (j & 3) * 2 + (j >> 2) * 16 + kh * 8;
      v2f f = __builtin_nontemporal_load((const v2f*)(arow + e));
      A1[kb].u[2 * j]     = f2bf(f[0]);
      A1[kb].u[2 * j + 1] = f2bf(f[1]);
    }
  }

  // ---- GEMM1: h = relu(x @ W1^T + b1)  (16xE)x(ExH) -----------------------
  unsigned short* hrow = Hlds + w * (16 * H_DIM);
  #pragma unroll 1
  for (int nt = 0; nt < 16; ++nt) {
    const int ncol = nt * 16 + m;
    const float bv = b1[ncol];
    v8f acc = { bv, bv, bv, bv, bv, bv, bv, bv };
    #pragma unroll
    for (int kb = 0; kb < 4; ++kb) {
      Frag B = load_bfrag_global(wbf + W1_OFF, ncol, E_DIM, kb * 32, kh);
      acc = __builtin_amdgcn_wmma_f32_16x16x32_bf16(
          false, A1[kb].v, false, B.v, (short)0, acc, false, false);
    }
    #pragma unroll
    for (int v = 0; v < 8; ++v) {
      float r = fmaxf(acc[v], 0.0f);                 // ReLU -> v_max_num_f32
      hrow[(v + kh * 8) * H_DIM + ncol] = f2bf(r);
    }
  }
  __syncthreads();

  // ---- GEMM2: x2 = h @ W2^T + b2  (16xH)x(HxE) ----------------------------
  Frag A2[8];
  #pragma unroll
  for (int kb = 0; kb < 8; ++kb)
    A2[kb] = load_afrag_lds(hrow + m * H_DIM + kb * 32 + kh * 8);

  unsigned short* x2row = X2lds + w * (16 * E_DIM);
  #pragma unroll 1
  for (int nt = 0; nt < 8; ++nt) {
    const int ncol = nt * 16 + m;                    // e index
    const float bv = b2[ncol];
    v8f acc = { bv, bv, bv, bv, bv, bv, bv, bv };
    #pragma unroll
    for (int kb = 0; kb < 8; ++kb) {
      Frag B = load_bfrag_global(wbf + W2_OFF, ncol, H_DIM, kb * 32, kh);
      acc = __builtin_amdgcn_wmma_f32_16x16x32_bf16(
          false, A2[kb].v, false, B.v, (short)0, acc, false, false);
    }
    #pragma unroll
    for (int v = 0; v < 8; ++v)
      x2row[(v + kh * 8) * E_DIM + ncol] = f2bf(acc[v]);
  }
  __syncthreads();

  // ---- GEMM3: s = x2 @ Winv^T + binv  (16xE)x(Ex64) + overlap-add ---------
  Frag A3[4];
  #pragma unroll
  for (int kb = 0; kb < 4; ++kb)
    A3[kb] = load_afrag_lds(x2row + m * E_DIM + kb * 32 + kh * 8);

  #pragma unroll 1
  for (int nt = 0; nt < 4; ++nt) {
    const int s = nt * 16 + m;                       // position within segment
    const float bv = binv[s];
    v8f acc = { bv, bv, bv, bv, bv, bv, bv, bv };
    #pragma unroll
    for (int kb = 0; kb < 4; ++kb) {
      Frag B = load_bfrag_global(wbf + WINV_OFF, s, E_DIM, kb * 32, kh);
      acc = __builtin_amdgcn_wmma_f32_16x16x32_bf16(
          false, A3[kb].v, false, B.v, (short)0, acc, false, false);
    }
    #pragma unroll
    for (int v = 0; v < 8; ++v) {
      int tseg = w * 16 + v + kh * 8;                // segment index 0..31
      atomicAdd(&sig[tseg * STEP + s], acc[v]);      // ds_add_f32 overlap-add
    }
  }
  __syncthreads();

  // ---- normalize by coverage count; keep in L2 for conv_merge -------------
  float* outp = gsig + (size_t)bc * SIG;
  for (int p = threadIdx.x; p < SIG; p += 64) {
    float scale = (p < STEP || p >= SIG - STEP) ? 1.0f : 0.5f;
    outp[p] = sig[p] * scale;
  }
}

// ---------------------------------------------------------------------------
// Kernel 2: ChannelMerger Conv1d(C=8 -> 1, k=3, pad=1) per batch
// ---------------------------------------------------------------------------
__global__ __launch_bounds__(256) void conv_merge(
    const float* __restrict__ gsig, const float* __restrict__ Wconv,
    const float* __restrict__ bconv, float* __restrict__ out)
{
  __shared__ float wc[C_CH * 3];
  __shared__ float bb;
  if (threadIdx.x < C_CH * 3) wc[threadIdx.x] = Wconv[threadIdx.x];
  if (threadIdx.x == 0)       bb = bconv[0];
  __syncthreads();

  const int b = blockIdx.x;
  const float* xb = gsig + (size_t)b * C_CH * SIG;

  for (int p = threadIdx.x; p < SIG; p += blockDim.x) {
    float acc = bb;
    #pragma unroll
    for (int c = 0; c < C_CH; ++c) {
      const float* xc = xb + c * SIG;
      float xm = (p > 0)       ? xc[p - 1] : 0.0f;
      float x0 = xc[p];
      float xp = (p < SIG - 1) ? xc[p + 1] : 0.0f;
      acc += wc[c * 3 + 0] * xm + wc[c * 3 + 1] * x0 + wc[c * 3 + 2] * xp;
    }
    // final output is never re-read on device -> non-temporal store
    __builtin_nontemporal_store(acc, out + (size_t)b * SIG + p);
  }
}

// ---------------------------------------------------------------------------
// Launcher
// ---------------------------------------------------------------------------
extern "C" void kernel_launch(void* const* d_in, const int* in_sizes, int n_in,
                              void* d_out, int out_size, void* d_ws, size_t ws_size,
                              hipStream_t stream)
{
  const float* enc   = (const float*)d_in[0];   // (1024, 8, 32, 128)
  const float* W1    = (const float*)d_in[1];   // (256, 128)
  const float* b1    = (const float*)d_in[2];   // (256,)
  const float* W2    = (const float*)d_in[3];   // (128, 256)
  const float* b2    = (const float*)d_in[4];   // (128,)
  const float* Winv  = (const float*)d_in[5];   // (64, 128)
  const float* binv  = (const float*)d_in[6];   // (64,)
  const float* Wconv = (const float*)d_in[7];   // (1, 8, 3)
  const float* bconv = (const float*)d_in[8];   // (1,)

  unsigned short* wbf  = (unsigned short*)d_ws;
  float*          gsig = (float*)((char*)d_ws + GSIG_BYTE_OFF); // 1024*8*1056 f32

  convert_weights<<<(WB_TOTAL + 255) / 256, 256, 0, stream>>>(W1, W2, Winv, wbf);
  decoder_gemm_ola<<<B_BATCH * C_CH, 64, 0, stream>>>(enc, b1, b2, binv, wbf, gsig);
  conv_merge<<<B_BATCH, 256, 0, stream>>>(gsig, Wconv, bconv, (float*)d_out);
}
